// RNNDecoder_53343493816552
// MI455X (gfx1250) — matrix operand order repaired and checked by reading, not compiled, not run on hardware
//
#include <hip/hip_runtime.h>
#include <stdint.h>

// Problem dimensions (compile-time, from the reference).
#define B_ 128
#define S_ 32
#define V_ 32000
#define E_ 512
#define H_ 1024
#define T_ 32

typedef __bf16 bf16_t;
typedef __attribute__((ext_vector_type(16))) __bf16   v16bf;
typedef __attribute__((ext_vector_type(8)))  float    v8f;
typedef __attribute__((ext_vector_type(4)))  uint32_t u32x4;

#define LDS_AS __attribute__((address_space(3)))
#define GLB_AS __attribute__((address_space(1)))

union Frag { v16bf v; u32x4 q[2]; };

__device__ __forceinline__ bf16_t f2bf(float f) {
  union { float f; uint32_t u; } in; in.f = f;
  uint32_t r = in.u + 0x7FFFu + ((in.u >> 16) & 1u);   // round-to-nearest-even
  union { unsigned short s; bf16_t b; } out; out.s = (unsigned short)(r >> 16);
  return out.b;
}

__device__ __forceinline__ float sigmoidf_(float x) {
  return 1.0f / (1.0f + __expf(-x));
}

// Async global->LDS copy (ASYNCcnt-tracked) with synchronous fallback.
// Builtin signature (from hipcc diagnostic): (v4i __device__*, v4i LDS*, imm, imm).
__device__ __forceinline__ void async_copy_b128(const bf16_t* g, bf16_t* l) {
#if __has_builtin(__builtin_amdgcn_global_load_async_to_lds_b128)
  typedef int v4i_ __attribute__((vector_size(16)));
  bf16_t* gnc = const_cast<bf16_t*>(g);
  __builtin_amdgcn_global_load_async_to_lds_b128(
      (GLB_AS v4i_*)gnc, (LDS_AS v4i_*)l, 0, 0);
#else
  *(u32x4*)l = *(const u32x4*)g;
#endif
}

template <int N>
__device__ __forceinline__ void wait_asynccnt() {
#if __has_builtin(__builtin_amdgcn_s_wait_asynccnt)
  __builtin_amdgcn_s_wait_asynccnt(N);
#else
  asm volatile("s_wait_asynccnt %0" :: "i"(N) : "memory");
#endif
}

// ---------------------------------------------------------------------------
// One-time f32 -> bf16 weight conversion.
// ---------------------------------------------------------------------------
__global__ void cvt_bf16_kernel(const float* __restrict__ in,
                                bf16_t* __restrict__ out, int n) {
  int i = blockIdx.x * blockDim.x + threadIdx.x;
  if (i < n) out[i] = f2bf(in[i]);
}

// ---------------------------------------------------------------------------
// Embedding gather: toks = [SOS=1, x[:,1..T-1]]; xs[t][b][:] = bf16(emb[tok]).
// ---------------------------------------------------------------------------
__global__ void embed_kernel(const int* __restrict__ x,
                             const float* __restrict__ emb,
                             bf16_t* __restrict__ xs) {
  int t = blockIdx.x / B_;
  int b = blockIdx.x % B_;
  int tok = (t == 0) ? 1 : x[b * S_ + t];
  const float* row = emb + (size_t)tok * E_;
  bf16_t* orow = xs + ((size_t)t * B_ + b) * E_;
  for (int i = threadIdx.x; i < E_; i += blockDim.x) orow[i] = f2bf(row[i]);
}

// ---------------------------------------------------------------------------
// Hidden-state init from encoder_hidden_state [L=2, B, H] (f32 + bf16 copy).
// ---------------------------------------------------------------------------
__global__ void init_h_kernel(const float* __restrict__ ehs,
                              float* __restrict__ h0f, float* __restrict__ h1f,
                              bf16_t* __restrict__ h0b, bf16_t* __restrict__ h1b) {
  int i = blockIdx.x * blockDim.x + threadIdx.x;
  if (i < B_ * H_) {
    float a = ehs[i];
    float c = ehs[B_ * H_ + i];
    h0f[i] = a; h0b[i] = f2bf(a);
    h1f[i] = c; h1b[i] = f2bf(c);
  }
}

// ---------------------------------------------------------------------------
// Fused GRU cell. One wave owns a 16x16 tile of h_new:
//   6 WMMA accumulation chains (gi_r/z/n over KIN, gh_r/z/n over H),
//   then gate math + state update. Grid: 128 blocks x 128 threads (512 waves
//   = 8 row-tiles x 64 col-tiles). Latency-bound (serial dependency chain),
//   so kept simple: direct global b128 fragment loads.
// ---------------------------------------------------------------------------
template <int KIN>
__global__ __launch_bounds__(128) void gru_cell_kernel(
    const bf16_t* __restrict__ xin,      // [B, KIN]
    const bf16_t* __restrict__ w_ih,     // [3H, KIN]
    const bf16_t* __restrict__ w_hh,     // [3H, H]
    const float*  __restrict__ b_ih,     // [3H]
    const float*  __restrict__ b_hh,     // [3H]
    const bf16_t* __restrict__ h_in_bf,  // [B, H]
    float*        __restrict__ h_f32,    // [B, H] in/out
    bf16_t*       __restrict__ h_out_bf, // [B, H]
    bf16_t*       __restrict__ y_out_bf) // optional extra sink (ys[t]) or null
{
  const int wid  = threadIdx.x >> 5;
  const int lane = threadIdx.x & 31;
  const int tile = blockIdx.x * 4 + wid;
  const int m0   = (tile & 7) * 16;   // batch-row tile (B/16 = 8)
  const int n0   = (tile >> 3) * 16;  // hidden-col tile (H/16 = 64)
  const int half = lane >> 4;
  const int lr   = lane & 15;

  v8f acc[6];
#pragma unroll
  for (int g = 0; g < 6; ++g)
#pragma unroll
    for (int i = 0; i < 8; ++i) acc[g][i] = 0.0f;

  // gi = x @ w_ih^T   (three gate column-blocks share the A fragment)
  {
    const bf16_t* arow = xin + (size_t)(m0 + lr) * KIN;
#pragma unroll 2
    for (int k0 = 0; k0 < KIN; k0 += 32) {
      Frag a;
      a.q[0] = *(const u32x4*)(arow + k0 + 16 * half);
      a.q[1] = *(const u32x4*)(arow + k0 + 16 * half + 8);
#pragma unroll
      for (int g = 0; g < 3; ++g) {
        const bf16_t* brow = w_ih + (size_t)(g * H_ + n0 + lr) * KIN;
        Frag b;
        b.q[0] = *(const u32x4*)(brow + k0 + 16 * half);
        b.q[1] = *(const u32x4*)(brow + k0 + 16 * half + 8);
        acc[g] = __builtin_amdgcn_wmma_f32_16x16x32_bf16(
            false, a.v, false, b.v, (short)0, acc[g], false, false);
      }
    }
  }
  // gh = h @ w_hh^T
  {
    const bf16_t* arow = h_in_bf + (size_t)(m0 + lr) * H_;
#pragma unroll 2
    for (int k0 = 0; k0 < H_; k0 += 32) {
      Frag a;
      a.q[0] = *(const u32x4*)(arow + k0 + 16 * half);
      a.q[1] = *(const u32x4*)(arow + k0 + 16 * half + 8);
#pragma unroll
      for (int g = 0; g < 3; ++g) {
        const bf16_t* brow = w_hh + (size_t)(g * H_ + n0 + lr) * H_;
        Frag b;
        b.q[0] = *(const u32x4*)(brow + k0 + 16 * half);
        b.q[1] = *(const u32x4*)(brow + k0 + 16 * half + 8);
        acc[3 + g] = __builtin_amdgcn_wmma_f32_16x16x32_bf16(
            false, a.v, false, b.v, (short)0, acc[3 + g], false, false);
      }
    }
  }

  // Gate math + state update. C/D layout: VGPR p -> M = p + 8*half, N = lr.
  const int col = n0 + lr;
  const float bir = b_ih[col],          bhr = b_hh[col];
  const float biz = b_ih[H_ + col],     bhz = b_hh[H_ + col];
  const float bin = b_ih[2 * H_ + col], bhn = b_hh[2 * H_ + col];
#pragma unroll
  for (int p = 0; p < 8; ++p) {
    const int row = m0 + p + 8 * half;
    const size_t idx = (size_t)row * H_ + col;
    float r = sigmoidf_((acc[0][p] + bir) + (acc[3][p] + bhr));
    float z = sigmoidf_((acc[1][p] + biz) + (acc[4][p] + bhz));
    float n = tanhf((acc[2][p] + bin) + r * (acc[5][p] + bhn));
    float hnew = (1.0f - z) * n + z * h_f32[idx];
    h_f32[idx] = hnew;
    bf16_t hb = f2bf(hnew);
    h_out_bf[idx] = hb;
    if (y_out_bf) y_out_bf[idx] = hb;
  }
}

// ---------------------------------------------------------------------------
// Vocab projection: logits[t*B+b][v] = ys[t*B+b][:] . w_dec[v][:] + b_dec[v].
//
// Block (4 waves, 128 thr) computes a 128(M) x 64(N) macro-tile; each wave
// owns 32(M) x 64(N) = 2x4 WMMA tiles. The shared 32(K) x 64(N) w_dec tile
// (4 KB) is staged into LDS with GLOBAL_LOAD_ASYNC_TO_LDS_B128, double
// buffered so the async copy of step k+1 overlaps the 8 WMMAs of step k
// (s_wait_asynccnt 2 -> previous buffer complete, in-order async loads).
// w_dec (65 MB bf16) is L2-resident (192 MB L2); the 524 MB f32 logits
// stream uses non-temporal stores so it does not evict it.
// ---------------------------------------------------------------------------
__global__ __launch_bounds__(128) void decoder_proj_kernel(
    const bf16_t* __restrict__ ys,   // [T*B, H]
    const bf16_t* __restrict__ w,    // [V, H]
    const float*  __restrict__ bias, // [V]
    float*        __restrict__ out)  // [T*B, V]
{
  __shared__ __attribute__((aligned(16))) bf16_t btile[2][64 * 32];  // 2 x 4KB

  const int tid  = threadIdx.x;
  const int wid  = tid >> 5;
  const int lane = tid & 31;
  const int half = lane >> 4;
  const int lr   = lane & 15;

  const int nt = blockIdx.x % (V_ / 64);       // 500 col blocks of 64
  const int mb = blockIdx.x / (V_ / 64);       // 32 row blocks of 128
  const int n0 = nt * 64;
  const int m0 = mb * 128 + wid * 32;          // this wave's 32 rows

  // Stage the 64x32 bf16 B tile (4KB) as 256 16-byte chunks, 2 per thread.
  auto stage = [&](bf16_t* ltile, int k0) {
#pragma unroll
    for (int c = tid; c < 256; c += 128) {
      const int r  = c >> 2;         // row (n) 0..63
      const int kk = (c & 3) * 8;    // K element offset 0,8,16,24
      async_copy_b128(w + (size_t)(n0 + r) * H_ + k0 + kk,
                      ltile + r * 32 + kk);
    }
  };

  v8f acc[2][4];
#pragma unroll
  for (int mi = 0; mi < 2; ++mi)
#pragma unroll
    for (int nf = 0; nf < 4; ++nf)
#pragma unroll
      for (int i = 0; i < 8; ++i) acc[mi][nf][i] = 0.0f;

  const bf16_t* arow0 = ys + (size_t)(m0 + lr) * H_;
  const bf16_t* arow1 = ys + (size_t)(m0 + 16 + lr) * H_;

  auto compute = [&](const bf16_t* lt, int k0) {
    Frag a[2], b[4];
    a[0].q[0] = *(const u32x4*)(arow0 + k0 + 16 * half);
    a[0].q[1] = *(const u32x4*)(arow0 + k0 + 16 * half + 8);
    a[1].q[0] = *(const u32x4*)(arow1 + k0 + 16 * half);
    a[1].q[1] = *(const u32x4*)(arow1 + k0 + 16 * half + 8);
#pragma unroll
    for (int nf = 0; nf < 4; ++nf) {
      const bf16_t* lrow = lt + (nf * 16 + lr) * 32 + 16 * half;
      b[nf].q[0] = *(const u32x4*)(lrow);
      b[nf].q[1] = *(const u32x4*)(lrow + 8);
    }
#pragma unroll
    for (int mi = 0; mi < 2; ++mi)
#pragma unroll
      for (int nf = 0; nf < 4; ++nf)
        acc[mi][nf] = __builtin_amdgcn_wmma_f32_16x16x32_bf16(
            false, a[mi].v, false, b[nf].v, (short)0, acc[mi][nf], false, false);
  };

  // Double-buffered K loop.
  stage(btile[0], 0);
  int buf = 0;
#pragma unroll 2
  for (int k0 = 0; k0 < H_ - 32; k0 += 32) {
    stage(btile[buf ^ 1], k0 + 32);   // prefetch next B tile (async)
    wait_asynccnt<2>();               // own copies for current tile done
    __syncthreads();                  // whole block's copies visible
    compute(btile[buf], k0);
    __syncthreads();                  // done reading before restage
    buf ^= 1;
  }
  wait_asynccnt<0>();
  __syncthreads();
  compute(btile[buf], H_ - 32);

  // Epilogue: bias + non-temporal f32 stores.
#pragma unroll
  for (int nf = 0; nf < 4; ++nf) {
    const int col = n0 + nf * 16 + lr;
    const float bb = bias[col];
#pragma unroll
    for (int mi = 0; mi < 2; ++mi) {
#pragma unroll
      for (int p = 0; p < 8; ++p) {
        const int row = m0 + mi * 16 + p + 8 * half;
        __builtin_nontemporal_store(acc[mi][nf][p] + bb,
                                    out + (size_t)row * V_ + col);
      }
    }
  }
}

// ---------------------------------------------------------------------------
// Host orchestration.
// ---------------------------------------------------------------------------
extern "C" void kernel_launch(void* const* d_in, const int* in_sizes, int n_in,
                              void* d_out, int out_size, void* d_ws, size_t ws_size,
                              hipStream_t stream) {
  (void)in_sizes; (void)n_in; (void)out_size; (void)ws_size;
  const int*   x     = (const int*)d_in[0];
  const float* ehs   = (const float*)d_in[2];
  const float* emb   = (const float*)d_in[3];
  const float* w_ih0 = (const float*)d_in[4];
  const float* w_hh0 = (const float*)d_in[5];
  const float* b_ih0 = (const float*)d_in[6];
  const float* b_hh0 = (const float*)d_in[7];
  const float* w_ih1 = (const float*)d_in[8];
  const float* w_hh1 = (const float*)d_in[9];
  const float* b_ih1 = (const float*)d_in[10];
  const float* b_hh1 = (const float*)d_in[11];
  const float* w_dec = (const float*)d_in[12];
  const float* b_dec = (const float*)d_in[13];
  float* out = (float*)d_out;

  // Workspace carve-up (~102 MB total).
  char* ws = (char*)d_ws;
  size_t off = 0;
  auto take = [&](size_t bytes) -> char* {
    char* p = ws + off;
    off += (bytes + 255) & ~(size_t)255;
    return p;
  };
  bf16_t* w_ih0_b = (bf16_t*)take((size_t)3 * H_ * E_ * 2);
  bf16_t* w_hh0_b = (bf16_t*)take((size_t)3 * H_ * H_ * 2);
  bf16_t* w_ih1_b = (bf16_t*)take((size_t)3 * H_ * H_ * 2);
  bf16_t* w_hh1_b = (bf16_t*)take((size_t)3 * H_ * H_ * 2);
  bf16_t* w_dec_b = (bf16_t*)take((size_t)V_ * H_ * 2);
  bf16_t* xs      = (bf16_t*)take((size_t)T_ * B_ * E_ * 2);
  bf16_t* ysbuf   = (bf16_t*)take((size_t)T_ * B_ * H_ * 2);
  float*  h0f     = (float*)take((size_t)B_ * H_ * 4);
  float*  h1f     = (float*)take((size_t)B_ * H_ * 4);
  bf16_t* h0b[2]  = {(bf16_t*)take((size_t)B_ * H_ * 2),
                     (bf16_t*)take((size_t)B_ * H_ * 2)};
  bf16_t* h1b[2]  = {(bf16_t*)take((size_t)B_ * H_ * 2),
                     (bf16_t*)take((size_t)B_ * H_ * 2)};

  auto cvt = [&](const float* src, bf16_t* dst, int n) {
    cvt_bf16_kernel<<<(n + 255) / 256, 256, 0, stream>>>(src, dst, n);
  };
  cvt(w_ih0, w_ih0_b, 3 * H_ * E_);
  cvt(w_hh0, w_hh0_b, 3 * H_ * H_);
  cvt(w_ih1, w_ih1_b, 3 * H_ * H_);
  cvt(w_hh1, w_hh1_b, 3 * H_ * H_);
  cvt(w_dec, w_dec_b, V_ * H_);

  embed_kernel<<<T_ * B_, 128, 0, stream>>>(x, emb, xs);
  init_h_kernel<<<(B_ * H_ + 255) / 256, 256, 0, stream>>>(ehs, h0f, h1f,
                                                           h0b[0], h1b[0]);

  // Sequential GRU over T steps; bf16 hidden copies ping-pong per layer.
  for (int t = 0; t < T_; ++t) {
    const int p = t & 1;
    gru_cell_kernel<E_><<<128, 128, 0, stream>>>(
        xs + (size_t)t * B_ * E_, w_ih0_b, w_hh0_b, b_ih0, b_hh0,
        h0b[p], h0f, h0b[1 - p], (bf16_t*)nullptr);
    gru_cell_kernel<H_><<<128, 128, 0, stream>>>(
        h0b[1 - p], w_ih1_b, w_hh1_b, b_ih1, b_hh1,
        h1b[p], h1f, h1b[1 - p], ysbuf + (size_t)t * B_ * H_);
  }

  // Projection: 32 row-blocks x 500 col-blocks = 16000 blocks, 4 waves each.
  decoder_proj_kernel<<<32 * (V_ / 64), 128, 0, stream>>>(ysbuf, w_dec_b,
                                                          b_dec, out);
}
